// B_Cell_35390530519886
// MI455X (gfx1250) — matrix operand order repaired and checked by reading, not compiled, not run on hardware
//
#include <hip/hip_runtime.h>
#include <hip/hip_bf16.h>
#include <math.h>

#define BATCH 65536
#define IDIM  256
#define HDIM  256
#define BDIM  128
#define LD    40   // LDS row stride in halves: 32 + 8 pad (80B -> conflict-light)

typedef __attribute__((ext_vector_type(16))) _Float16 v16h;
typedef __attribute__((ext_vector_type(8)))  _Float16 v8h;
typedef __attribute__((ext_vector_type(4)))  _Float16 v4h;
typedef __attribute__((ext_vector_type(8)))  float    v8f;

__device__ __forceinline__ v16h cat8(v8h lo, v8h hi) {
  return __builtin_shufflevector(lo, hi, 0,1,2,3,4,5,6,7,8,9,10,11,12,13,14,15);
}

// A fragment (16x32 f16, MxK). lane: half=l>>4, row=l&15 (caller passes row ptr).
// elems 0-7 = K[half*8 .. +7], elems 8-15 = K[16+half*8 .. +7]  (ISA 7.12.2)
__device__ __forceinline__ v16h frag_a(const _Float16* row, int half) {
  v8h lo = *(const v8h*)(row + half * 8);
  v8h hi = *(const v8h*)(row + 16 + half * 8);
  return cat8(lo, hi);
}

// B fragment (32x16 f16, KxN) from LDS staged [n][k]-major:
// lane col=l&15, elems 0-7 = K[half*8 .. +7], elems 8-15 = K[16+half*8 .. +7].
__device__ __forceinline__ v16h frag_b(const _Float16* row, int half) {
  v8h lo = *(const v8h*)(row + half * 8);
  v8h hi = *(const v8h*)(row + 16 + half * 8);
  return cat8(lo, hi);
}

__device__ __forceinline__ v8f wmma16(v16h a, v16h b, v8f c) {
  return __builtin_amdgcn_wmma_f32_16x16x32_f16(false, a, false, b,
                                                (short)0, c, false, false);
}

// stage a ROWS x 32 f32 tile -> f16 LDS [ROWS][LD], f32->f16 convert inline
template<int ROWS>
__device__ __forceinline__ void stage(const float* __restrict__ src, int src_stride,
                                      _Float16* __restrict__ dst, int tid) {
  constexpr int chunks = ROWS * 8;  // float4 chunks (8 per 32-wide row)
#pragma unroll
  for (int i0 = 0; i0 < chunks; i0 += 256) {
    int i = i0 + tid;
    int r = i >> 3, c = (i & 7) * 4;
    float4 v = *(const float4*)(src + (size_t)r * src_stride + c);
    v4h h = { (_Float16)v.x, (_Float16)v.y, (_Float16)v.z, (_Float16)v.w };
    *(v4h*)(dst + r * LD + c) = h;
  }
}

// out[m][n] = tanh(X@Wa^T + ba)  [+ tanh(Y@Wh^T + X@Pt + bh)]
// Pt already carries the -1/||v||^2 sign/scale, so the projection is pure
// accumulation (no NEG modifier needed; f16 WMMA only allows CNeg per ISA).
template<int NOUT, bool HASY, bool HASP>
__global__ __launch_bounds__(256)
void cell_kernel(const float* __restrict__ X, const float* __restrict__ Wa,
                 const float* __restrict__ ba,
                 const float* __restrict__ Y, const float* __restrict__ Wh,
                 const float* __restrict__ bh,
                 const float* __restrict__ Pt,
                 float* __restrict__ out) {
  __shared__ _Float16 sX[64 * LD], sY[64 * LD];
  __shared__ _Float16 sWa[128 * LD], sWh[128 * LD], sP[128 * LD];
  const int tid = threadIdx.x;
  const int w = tid >> 5, lane = tid & 31;
  const int wm = w >> 1, wn = w & 1;          // 4 waves along M, 2 along N
  const int half = lane >> 4, r16 = lane & 15;
  const int mblock = blockIdx.x * 64;
  const int nblock = blockIdx.y * 128;

  const v8f vz = {0.f,0.f,0.f,0.f,0.f,0.f,0.f,0.f};
  v8f acc1[4] = {vz, vz, vz, vz};
  v8f acc2[4] = {vz, vz, vz, vz};

  for (int k0 = 0; k0 < 256; k0 += 32) {
    stage<64>(X + (size_t)mblock * 256 + k0, 256, sX, tid);
    stage<128>(Wa + (size_t)nblock * 256 + k0, 256, sWa, tid);
    if (HASY) {
      stage<64>(Y + (size_t)mblock * 256 + k0, 256, sY, tid);
      stage<128>(Wh + (size_t)nblock * 256 + k0, 256, sWh, tid);
    }
    if (HASP) stage<128>(Pt + (size_t)nblock * 256 + k0, 256, sP, tid);
    __syncthreads();

    const v16h aX = frag_a(sX + (wm * 16 + r16) * LD, half);
    v16h aY;
    if (HASY) aY = frag_a(sY + (wm * 16 + r16) * LD, half);

#pragma unroll
    for (int t = 0; t < 4; ++t) {
      const int nrow = (wn * 64 + t * 16 + r16) * LD;
      acc1[t] = wmma16(aX, frag_b(sWa + nrow, half), acc1[t]);
      if (HASY) acc2[t] = wmma16(aY, frag_b(sWh + nrow, half), acc2[t]);
      if (HASP) acc2[t] = wmma16(aX, frag_b(sP + nrow, half), acc2[t]);
    }
    __syncthreads();
  }

#pragma unroll
  for (int t = 0; t < 4; ++t) {
    const int gcol = nblock + wn * 64 + t * 16 + r16;
    const float b1 = ba[gcol];
    const float b2 = HASY ? bh[gcol] : 0.f;
#pragma unroll
    for (int v = 0; v < 8; ++v) {
      const int grow = mblock + wm * 16 + half * 8 + v;   // C layout: VGPR v, lane half
      float val = tanhf(acc1[t][v] + b1);
      if (HASY) val += tanhf(acc2[t][v] + b2);
      out[(size_t)grow * NOUT + gcol] = val;
    }
  }
}

// G[k][j] += sum_b A[b][k] * Hm[b][j]  (split-K over batch, f32 atomics)
__global__ __launch_bounds__(256)
void gram_kernel(const float* __restrict__ A, const float* __restrict__ Hm,
                 float* __restrict__ G) {
  __shared__ _Float16 sA[64 * LD], sB[128 * LD];
  const int tid = threadIdx.x;
  const int w = tid >> 5, lane = tid & 31;
  const int wm = w >> 1, wn = w & 1;
  const int half = lane >> 4, r16 = lane & 15;
  const int mblock = blockIdx.x * 64;     // k-rows of G
  const int nblock = blockIdx.y * 128;    // j-cols of G
  const int bbase  = blockIdx.z * 512;    // batch slice

  const v8f vz = {0.f,0.f,0.f,0.f,0.f,0.f,0.f,0.f};
  v8f acc[4] = {vz, vz, vz, vz};

  for (int kb = 0; kb < 512; kb += 32) {
    const int b0 = bbase + kb;
    // transpose-stage: sA[m][kk] = A[b0+kk][mblock+m]
#pragma unroll
    for (int i0 = 0; i0 < 32 * 16; i0 += 256) {
      int i = i0 + tid;
      int kk = i >> 4, m4 = (i & 15) * 4;
      float4 v = *(const float4*)(A + (size_t)(b0 + kk) * 256 + mblock + m4);
      sA[(m4 + 0) * LD + kk] = (_Float16)v.x;
      sA[(m4 + 1) * LD + kk] = (_Float16)v.y;
      sA[(m4 + 2) * LD + kk] = (_Float16)v.z;
      sA[(m4 + 3) * LD + kk] = (_Float16)v.w;
    }
    // transpose-stage: sB[n][kk] = Hm[b0+kk][nblock+n]
#pragma unroll
    for (int i0 = 0; i0 < 32 * 32; i0 += 256) {
      int i = i0 + tid;
      int kk = i >> 5, n4 = (i & 31) * 4;
      float4 v = *(const float4*)(Hm + (size_t)(b0 + kk) * 256 + nblock + n4);
      sB[(n4 + 0) * LD + kk] = (_Float16)v.x;
      sB[(n4 + 1) * LD + kk] = (_Float16)v.y;
      sB[(n4 + 2) * LD + kk] = (_Float16)v.z;
      sB[(n4 + 3) * LD + kk] = (_Float16)v.w;
    }
    __syncthreads();

    const v16h a = frag_a(sA + (wm * 16 + r16) * LD, half);
#pragma unroll
    for (int t = 0; t < 4; ++t)
      acc[t] = wmma16(a, frag_b(sB + (wn * 64 + t * 16 + r16) * LD, half), acc[t]);
    __syncthreads();
  }

#pragma unroll
  for (int t = 0; t < 4; ++t) {
    const int gcol = nblock + wn * 64 + t * 16 + r16;
#pragma unroll
    for (int v = 0; v < 8; ++v) {
      const int grow = mblock + wm * 16 + half * 8 + v;
      atomicAdd(&G[(size_t)grow * 256 + gcol], acc[t][v]);
    }
  }
}

// Pt[n][k] = -(1/sumsq) * sum_j G[k][j] * Wh[n][j]   (tiny 256^3 GEMM, L2-resident)
__global__ __launch_bounds__(256)
void pmat_kernel(const float* __restrict__ G, const float* __restrict__ Wh,
                 const float* __restrict__ sumsq, float* __restrict__ Pt) {
  const int n = blockIdx.x, k = threadIdx.x;
  const float inv = 1.0f / *sumsq;
  const float* g  = G  + (size_t)k * 256;
  const float* wh = Wh + (size_t)n * 256;
  float s = 0.f;
#pragma unroll 8
  for (int j = 0; j < 256; ++j) s = fmaf(g[j], wh[j], s);
  Pt[(size_t)n * 256 + k] = -inv * s;
}

// sum of squares of two 16.7M-element arrays -> out[0], out[1]
__global__ __launch_bounds__(256)
void sumsq_kernel(const float* __restrict__ a, const float* __restrict__ b,
                  float* __restrict__ out, long n) {
  __shared__ float red0[256], red1[256];
  const int tid = threadIdx.x;
  float sa = 0.f, sb = 0.f;
  for (long i = (long)blockIdx.x * blockDim.x + tid; i < n;
       i += (long)gridDim.x * blockDim.x) {
    float x = a[i]; sa = fmaf(x, x, sa);
    float y = b[i]; sb = fmaf(y, y, sb);
  }
  red0[tid] = sa; red1[tid] = sb;
  __syncthreads();
  for (int s = 128; s > 0; s >>= 1) {
    if (tid < s) { red0[tid] += red0[tid + s]; red1[tid] += red1[tid + s]; }
    __syncthreads();
  }
  if (tid == 0) { atomicAdd(out, red0[0]); atomicAdd(out + 1, red1[0]); }
}

extern "C" void kernel_launch(void* const* d_in, const int* in_sizes, int n_in,
                              void* d_out, int out_size, void* d_ws, size_t ws_size,
                              hipStream_t stream) {
  (void)in_sizes; (void)n_in; (void)out_size; (void)ws_size;

  const float* h_prev = (const float*)d_in[0];
  const float* n_h    = (const float*)d_in[1];
  const float* n_r    = (const float*)d_in[2];
  const float* n_t    = (const float*)d_in[3];
  const float* W_nh   = (const float*)d_in[4];
  const float* b_nh   = (const float*)d_in[5];
  const float* W_nr   = (const float*)d_in[6];
  const float* b_nr   = (const float*)d_in[7];
  const float* W_nt   = (const float*)d_in[8];
  const float* b_nt   = (const float*)d_in[9];
  const float* W_bt   = (const float*)d_in[10];
  const float* b_bt   = (const float*)d_in[11];
  const float* W_h    = (const float*)d_in[12];
  const float* b_h    = (const float*)d_in[13];

  float* out    = (float*)d_out;
  float* h_next = out;                            // B x 256
  float* b_out  = out + (size_t)BATCH * HDIM;     // B x 128

  float* ws   = (float*)d_ws;
  float* h_i  = ws;                               // B x 256
  float* h_i1 = h_i  + (size_t)BATCH * HDIM;      // B x 256
  float* G1   = h_i1 + (size_t)BATCH * HDIM;      // 256 x 256
  float* G2   = G1 + 256 * 256;                   // 256 x 256
  float* sums = G2 + 256 * 256;                   // [||n_r||^2, ||n_t||^2]
  float* Pt1  = sums + 2;                         // 256 x 256
  float* Pt2  = Pt1 + 256 * 256;                  // 256 x 256

  // zero the atomic accumulators (G1, G2, sums are contiguous)
  hipMemsetAsync(G1, 0, (size_t)(2 * 256 * 256 + 2) * sizeof(float), stream);

  sumsq_kernel<<<512, 256, 0, stream>>>(n_r, n_t, sums, (long)BATCH * IDIM);

  dim3 blk(256);
  dim3 grid_h(BATCH / 64, HDIM / 128);
  dim3 grid_b(BATCH / 64, BDIM / 128);
  dim3 grid_g(IDIM / 64, HDIM / 128, 128);

  // h_i = tanh(n_h W_nh^T + b) + tanh(h_prev W_h^T + b)
  cell_kernel<HDIM, true, false><<<grid_h, blk, 0, stream>>>(
      n_h, W_nh, b_nh, h_prev, W_h, b_h, nullptr, h_i);
  // G1 = n_r^T h_i ;  Pt1 = -(W_h G1^T) / ||n_r||^2
  gram_kernel<<<grid_g, blk, 0, stream>>>(n_r, h_i, G1);
  pmat_kernel<<<256, 256, 0, stream>>>(G1, W_h, sums + 0, Pt1);
  // h_i_1 = tanh(n_r W_nr^T + b) + tanh(h_i W_h^T + n_r Pt1 + b)
  cell_kernel<HDIM, true, true><<<grid_h, blk, 0, stream>>>(
      n_r, W_nr, b_nr, h_i, W_h, b_h, Pt1, h_i1);
  // G2 = n_t^T h_i_1 ;  Pt2
  gram_kernel<<<grid_g, blk, 0, stream>>>(n_t, h_i1, G2);
  pmat_kernel<<<256, 256, 0, stream>>>(G2, W_h, sums + 1, Pt2);
  // h_next
  cell_kernel<HDIM, true, true><<<grid_h, blk, 0, stream>>>(
      n_t, W_nt, b_nt, h_i1, W_h, b_h, Pt2, h_next);
  // b = tanh(h_next W_bt^T + b_bt)
  cell_kernel<BDIM, false, false><<<grid_b, blk, 0, stream>>>(
      h_next, W_bt, b_bt, nullptr, nullptr, nullptr, nullptr, b_out);
}
